// NetCapClassifier_58445914964490
// MI455X (gfx1250) — matrix operand order repaired and checked by reading, not compiled, not run on hardware
//
#include <hip/hip_runtime.h>

// ---------------- problem constants (fixed by reference) ----------------
#define N0v 300000
#define N1v 150000
#define N2v 50000
#define E0v 2400000
#define E1v 800000
#define D_DEVv 12
#define D_INSTv 10
#define D_NETv 16
#define PROJv 56

typedef __attribute__((ext_vector_type(16))) _Float16 v16h;
typedef __attribute__((ext_vector_type(8)))  float    v8f;

// ---- f16 weight region: fragment-swizzled, 512 halves per 32x16 fragment ----
// Fragment order inside a matrix: f = kc * (NP/16) + nt  (kc = K/32 chunk, nt = N/16 tile)
// Inside a fragment: element i = lane*16 + e, holding W[kc*32 + (lane&16?8:0) + (e<8?e:e+8)][nt*16 + (lane&15)]
enum : int {
  OFF_WE1  = 0,                     // 384 x 16  -> 12 frags
  OFF_WE2  = OFF_WE1  + 384 * 16,   // 32 x 16   ->  1 frag (K 16->32, N 8->16)
  OFF_WDEV = OFF_WE2  + 32 * 16,    // 32 x 64   ->  4 frags (K 12->32, N 56->64)
  OFF_WINST= OFF_WDEV + 32 * 64,
  OFF_WNET = OFF_WINST+ 32 * 64,
  OFF_WS0  = OFF_WNET + 32 * 64,    // 64 x 64   ->  8 frags
  OFF_WN0  = OFF_WS0  + 64 * 64,
  OFF_WS1  = OFF_WN0  + 64 * 64,
  OFF_WN1  = OFF_WS1  + 64 * 64,
  OFF_WC1  = OFF_WN1  + 64 * 64,    // 96 x 64   -> 12 frags (K 80->96)
  OFF_WC2  = OFF_WC1  + 96 * 64,
  OFF_WC3  = OFF_WC2  + 64 * 64,    // 64 x 16   ->  2 frags (N 8->16)
  HTOT     = OFF_WC3  + 64 * 16
};
#define WH_BYTES 80896
#define FRAG 512   // halves per 32x16 fragment

// ---------------- WMMA fragment loaders (ISA 05_wmma layouts) ----------------
// A 16x32 f16: lane holds row m = lane&15; per lane-half K is in two runs of 8:
//   lane<16 : elements 0..7 -> K=0..7,   8..15 -> K=16..23
//   lane>=16: elements 0..7 -> K=8..15,  8..15 -> K=24..31
__device__ __forceinline__ v16h load_a_f32(const float* base, int ld, int kmax) {
  const int lane = threadIdx.x & 31;
  const int m = lane & 15;
  const int khi = (lane & 16) ? 8 : 0;
  const float* rp = base + m * ld + khi;
  float4 x0 = *(const float4*)(rp);
  float4 x1 = *(const float4*)(rp + 4);
  float4 x2 = make_float4(0.f, 0.f, 0.f, 0.f);
  float4 x3 = x2;
  if (khi + 16 < kmax) {            // kmax is 16 or 32 at every call site
    x2 = *(const float4*)(rp + 16);
    x3 = *(const float4*)(rp + 20);
  }
  v16h a;
  a[0]=(_Float16)x0.x; a[1]=(_Float16)x0.y; a[2]=(_Float16)x0.z; a[3]=(_Float16)x0.w;
  a[4]=(_Float16)x1.x; a[5]=(_Float16)x1.y; a[6]=(_Float16)x1.z; a[7]=(_Float16)x1.w;
  a[8]=(_Float16)x2.x; a[9]=(_Float16)x2.y; a[10]=(_Float16)x2.z; a[11]=(_Float16)x2.w;
  a[12]=(_Float16)x3.x; a[13]=(_Float16)x3.y; a[14]=(_Float16)x3.z; a[15]=(_Float16)x3.w;
  return a;
}
__device__ __forceinline__ v16h load_a_f32_scaled(const float* base, int ld, int kmax, float s) {
  const int lane = threadIdx.x & 31;
  const int m = lane & 15;
  const int khi = (lane & 16) ? 8 : 0;
  const float* rp = base + m * ld + khi;
  float4 x0 = *(const float4*)(rp);
  float4 x1 = *(const float4*)(rp + 4);
  float4 x2 = make_float4(0.f, 0.f, 0.f, 0.f);
  float4 x3 = x2;
  if (khi + 16 < kmax) {
    x2 = *(const float4*)(rp + 16);
    x3 = *(const float4*)(rp + 20);
  }
  v16h a;
  a[0]=(_Float16)(x0.x*s); a[1]=(_Float16)(x0.y*s); a[2]=(_Float16)(x0.z*s); a[3]=(_Float16)(x0.w*s);
  a[4]=(_Float16)(x1.x*s); a[5]=(_Float16)(x1.y*s); a[6]=(_Float16)(x1.z*s); a[7]=(_Float16)(x1.w*s);
  a[8]=(_Float16)(x2.x*s); a[9]=(_Float16)(x2.y*s); a[10]=(_Float16)(x2.z*s); a[11]=(_Float16)(x2.w*s);
  a[12]=(_Float16)(x3.x*s); a[13]=(_Float16)(x3.y*s); a[14]=(_Float16)(x3.z*s); a[15]=(_Float16)(x3.w*s);
  return a;
}
// B fragment: pre-swizzled, one 32-byte contiguous run per lane
__device__ __forceinline__ v16h load_b_frag(const _Float16* fragbase) {
  const int lane = threadIdx.x & 31;
  return *(const v16h*)(fragbase + lane * 16);
}
__device__ __forceinline__ v8f wmma16(v16h a, v16h b, v8f c) {
  return __builtin_amdgcn_wmma_f32_16x16x32_f16(false, a, false, b, (short)0, c, false, false);
}
#define WAIT_LDS() asm volatile("s_wait_dscnt 0" ::: "memory")
// XOR-butterfly within 16-lane groups via ds_swizzle (offset = xor<<10 | and 0x1F)
#define SWZ_ADD(v, imm) \
  (v) += __int_as_float(__builtin_amdgcn_ds_swizzle(__float_as_int(v), (imm)))

// ---------------- kernel 0: convert + pad + fragment-swizzle weights ----------------
__device__ __forceinline__ void conv_mat(const float* src, _Float16* dst,
                                         int srcK, int srcN, int KP, int NP, int t) {
  const int nfn = NP >> 4;        // N-tiles per K-chunk
  const int total = KP * NP;
  for (int i = t; i < total; i += 256) {
    const int f = i >> 9;         // fragment index
    const int e9 = i & 511;
    const int lane = e9 >> 4, e = e9 & 15;
    const int kc = f / nfn, nt = f % nfn;
    const int k = kc * 32 + ((lane & 16) ? 8 : 0) + ((e < 8) ? e : e + 8);
    const int n = nt * 16 + (lane & 15);
    dst[i] = (_Float16)((k < srcK && n < srcN) ? src[k * srcN + n] : 0.0f);
  }
}
__global__ void k_convert(const float* We1, const float* We2,
                          const float* Wdev, const float* Winst, const float* Wnet,
                          const float* Ws0, const float* Wn0,
                          const float* Ws1, const float* Wn1,
                          const float* Wc1, const float* Wc2, const float* Wc3,
                          _Float16* wh) {
  const int t = threadIdx.x;
  conv_mat(We1,  wh + OFF_WE1,  384, 16, 384, 16, t);
  conv_mat(We2,  wh + OFF_WE2,   16,  8,  32, 16, t);
  conv_mat(Wdev, wh + OFF_WDEV, D_DEVv,  PROJv, 32, 64, t);
  conv_mat(Winst,wh + OFF_WINST,D_INSTv, PROJv, 32, 64, t);
  conv_mat(Wnet, wh + OFF_WNET, D_NETv,  PROJv, 32, 64, t);
  conv_mat(Ws0,  wh + OFF_WS0,  64, 64, 64, 64, t);
  conv_mat(Wn0,  wh + OFF_WN0,  64, 64, 64, 64, t);
  conv_mat(Ws1,  wh + OFF_WS1,  64, 64, 64, 64, t);
  conv_mat(Wn1,  wh + OFF_WN1,  64, 64, 64, 64, t);
  conv_mat(Wc1,  wh + OFF_WC1,  80, 64, 96, 64, t);
  conv_mat(Wc2,  wh + OFF_WC2,  64, 64, 64, 64, t);
  conv_mat(Wc3,  wh + OFF_WC3,  64,  8, 64, 16, t);
}

// ---------------- kernel: zero a float region ----------------
__global__ void k_zero(float* p, int n) {
  int i = blockIdx.x * blockDim.x + threadIdx.x;
  const int stride = gridDim.x * blockDim.x;
  for (; i < n; i += stride) p[i] = 0.0f;
}

// ---------------- kernel 1: fused node features -> h (N0 x 64) ----------------
__global__ void __launch_bounds__(256)
k_node(const float* feats, const int* ntypes, const float* hemb,
       const float* be1, const float* be2,
       const float* bdev, const float* binst, const float* bnet,
       const _Float16* wh, float* hout) {
  __shared__ __align__(16) float eL[8][16 * 16];
  __shared__ __align__(16) float e2L[8][16 * 8];
  __shared__ int ntL[8][16];
  const int wv = threadIdx.x >> 5;
  const int lane = threadIdx.x & 31;
  const int tile = blockIdx.x * 8 + wv;
  if (tile * 16 >= N0v) return;
  const int row0 = tile * 16;
  const int m = lane & 15;
  const int hi8 = (lane & 16) ? 8 : 0;

  // e = relu(h_embd @ We1 + be1): K=384 in 12 chunks of 32
  v8f acc = {};
  const float* A0 = hemb + (size_t)row0 * 384;
#pragma unroll 1
  for (int kc2 = 0; kc2 < 12; ++kc2) {
    v16h a = load_a_f32(A0 + kc2 * 32, 384, 32);
    v16h b = load_b_frag(wh + OFF_WE1 + kc2 * FRAG);
    acc = wmma16(a, b, acc);
  }
  const float bias1 = be1[m];
  float ev[8];
#pragma unroll
  for (int r = 0; r < 8; ++r) ev[r] = fmaxf(acc[r] + bias1, 0.0f);
  // row L2 norm: XOR-reduce across the 16 columns (= lanes within each half)
#pragma unroll
  for (int r = 0; r < 8; ++r) {
    float s = ev[r] * ev[r];
    SWZ_ADD(s, 0x041F);  // xor 1
    SWZ_ADD(s, 0x081F);  // xor 2
    SWZ_ADD(s, 0x101F);  // xor 4
    SWZ_ADD(s, 0x201F);  // xor 8
    float inv = 1.0f / fmaxf(sqrtf(s), 1e-12f);
    ev[r] *= inv;
    eL[wv][(r + hi8) * 16 + m] = ev[r];
  }
  if (lane < 16) ntL[wv][lane] = ntypes[row0 + lane];
  WAIT_LDS();

  // e2 = e @ We2 + be2 (valid cols 0..7); K padded 16->32
  {
    v16h a = load_a_f32(&eL[wv][0], 16, 16);
    v16h b = load_b_frag(wh + OFF_WE2);
    v8f c = {};
    c = wmma16(a, b, c);
    if (m < 8) {
      float b2 = be2[m];
#pragma unroll
      for (int r = 0; r < 8; ++r) e2L[wv][(r + hi8) * 8 + m] = c[r] + b2;
    }
  }

  // typed projections: feats(16x16, K pad->32) x {Wdev,Winst,Wnet}(32x64)
  v16h af = load_a_f32(feats + (size_t)row0 * 16, 16, 16);
  v8f pd[4], pi[4], pn[4];
#pragma unroll
  for (int nt = 0; nt < 4; ++nt) {
    v8f z = {};
    pd[nt] = wmma16(af, load_b_frag(wh + OFF_WDEV  + nt * FRAG), z);
    pi[nt] = wmma16(af, load_b_frag(wh + OFF_WINST + nt * FRAG), z);
    pn[nt] = wmma16(af, load_b_frag(wh + OFF_WNET  + nt * FRAG), z);
  }
  WAIT_LDS();

  // h = [proj(56) | e2(8)]
#pragma unroll
  for (int nt = 0; nt < 4; ++nt) {
    const int col = nt * 16 + m;
#pragma unroll
    for (int r = 0; r < 8; ++r) {
      const int rl = r + hi8;
      float v;
      if (col < PROJv) {
        const int ty = ntL[wv][rl];
        float bb = (ty == 0) ? bdev[col] : (ty == 1) ? binst[col] : bnet[col];
        float pv = (ty == 0) ? pd[nt][r] : (ty == 1) ? pi[nt][r] : pn[nt][r];
        v = pv + bb;
      } else {
        v = e2L[wv][rl * 8 + (col - PROJv)];
      }
      hout[(size_t)(row0 + rl) * 64 + col] = v;
    }
  }
}

// ---------------- kernel: edge aggregation (segment sum + degree) ----------------
__global__ void __launch_bounds__(256)
k_agg(const int* src, const int* dst, const float* h, float* agg, float* deg, int nedges) {
  const int e = blockIdx.x * 256 + threadIdx.x;
  if (e >= nedges) return;
  const int s = src[e], d = dst[e];
  const float4* hs = (const float4*)(h + (size_t)s * 64);
  float* ad = agg + (size_t)d * 64;
  __builtin_prefetch(hs, 0, 3);
  atomicAdd(&deg[d], 1.0f);
#pragma unroll 4
  for (int q = 0; q < 16; ++q) {
    float4 v = hs[q];
    atomicAdd(&ad[q * 4 + 0], v.x);
    atomicAdd(&ad[q * 4 + 1], v.y);
    atomicAdd(&ad[q * 4 + 2], v.z);
    atomicAdd(&ad[q * 4 + 3], v.w);
  }
}

// ---------------- kernel: GNN layer  out = act(hin@Ws + (agg/deg)@Wn + b) ----------------
// Persistent: each wave processes GNN_TPW tiles; 16 weight B-fragments hoisted to regs.
#define GNN_TPW 4
__global__ void __launch_bounds__(256)
k_gnn(const float* hin, const float* agg, const float* deg,
      const _Float16* wh, int offWs, int offWn,
      const float* bias, float* hout, int nrows, int do_relu) {
  const int wv = threadIdx.x >> 5;
  const int lane = threadIdx.x & 31;
  const int tbase = (blockIdx.x * 8 + wv) * GNN_TPW;
  if (tbase * 16 >= nrows) return;
  const int m = lane & 15;
  const int hi8 = (lane & 16) ? 8 : 0;

  // hoist all weight fragments (loop-invariant across tiles)
  v16h Bs[2][4], Bn[2][4];
#pragma unroll
  for (int kc2 = 0; kc2 < 2; ++kc2)
#pragma unroll
    for (int nt = 0; nt < 4; ++nt) {
      Bs[kc2][nt] = load_b_frag(wh + offWs + (kc2 * 4 + nt) * FRAG);
      Bn[kc2][nt] = load_b_frag(wh + offWn + (kc2 * 4 + nt) * FRAG);
    }
  float bcol[4];
#pragma unroll
  for (int nt = 0; nt < 4; ++nt) bcol[nt] = bias[nt * 16 + m];

#pragma unroll 1
  for (int t = 0; t < GNN_TPW; ++t) {
    const int tile = tbase + t;
    if (tile * 16 >= nrows) break;
    const int row0 = tile * 16;
    const float inv = 1.0f / fmaxf(deg[row0 + m], 1.0f);
    v8f acc[4] = {};
    const float* A0 = hin + (size_t)row0 * 64;
    const float* G0 = agg + (size_t)row0 * 64;
#pragma unroll
    for (int kc2 = 0; kc2 < 2; ++kc2) {
      v16h a = load_a_f32(A0 + kc2 * 32, 64, 32);
      v16h g = load_a_f32_scaled(G0 + kc2 * 32, 64, 32, inv);
#pragma unroll
      for (int nt = 0; nt < 4; ++nt) {
        acc[nt] = wmma16(a, Bs[kc2][nt], acc[nt]);
        acc[nt] = wmma16(g, Bn[kc2][nt], acc[nt]);
      }
    }
#pragma unroll
    for (int nt = 0; nt < 4; ++nt) {
      const int col = nt * 16 + m;
#pragma unroll
      for (int r = 0; r < 8; ++r) {
        float v = acc[nt][r] + bcol[nt];
        if (do_relu) v = fmaxf(v, 0.0f);
        hout[(size_t)(row0 + r + hi8) * 64 + col] = v;
      }
    }
  }
}

// ---------------- kernel: classifier head over h2 ----------------
__global__ void __launch_bounds__(256)
k_cls(const float* h2, const float* feats, const _Float16* wh,
      const float* bc1, const float* bc2, const float* bc3, float* logits) {
  __shared__ __align__(16) float st[8][16 * 96];
  const int wv = threadIdx.x >> 5;
  const int lane = threadIdx.x & 31;
  const int tile = blockIdx.x * 8 + wv;
  if (tile * 16 >= N2v) return;
  const int row0 = tile * 16;
  const int m = lane & 15;
  const int hi8 = (lane & 16) ? 8 : 0;
  float* S = &st[wv][0];

  // stage net_h = [h2 (64) | feats[:,:16]] in LDS, ld=96 (cols 80..95 never read)
#pragma unroll
  for (int i = 0; i < 8; ++i) {                 // 16x64 h2 = 256 float4s
    const int i4 = lane * 8 + i;
    const int r = i4 >> 4, c4 = i4 & 15;
    *(float4*)(S + r * 96 + c4 * 4) = *(const float4*)(h2 + (size_t)(row0 + r) * 64 + c4 * 4);
  }
#pragma unroll
  for (int i = 0; i < 2; ++i) {                 // 16x16 feats = 64 float4s
    const int i4 = lane * 2 + i;
    const int r = i4 >> 2, c4 = i4 & 3;
    *(float4*)(S + r * 96 + 64 + c4 * 4) = *(const float4*)(feats + (size_t)(row0 + r) * 16 + c4 * 4);
  }
  WAIT_LDS();

  // x1 = relu(net_h @ Wc1 + bc1), K=80 padded->96 (last chunk kmax=16)
  v8f acc[4] = {};
#pragma unroll
  for (int kc2 = 0; kc2 < 3; ++kc2) {
    v16h a = load_a_f32(S + kc2 * 32, 96, (kc2 < 2) ? 32 : 16);
#pragma unroll
    for (int nt = 0; nt < 4; ++nt)
      acc[nt] = wmma16(a, load_b_frag(wh + OFF_WC1 + (kc2 * 4 + nt) * FRAG), acc[nt]);
  }
  WAIT_LDS();
#pragma unroll
  for (int nt = 0; nt < 4; ++nt) {
    const int col = nt * 16 + m;
    const float bb = bc1[col];
#pragma unroll
    for (int r = 0; r < 8; ++r)
      S[(r + hi8) * 96 + col] = fmaxf(acc[nt][r] + bb, 0.0f);
  }
  WAIT_LDS();

  // x2 = relu(x1 @ Wc2 + bc2)
  v8f acc2[4] = {};
#pragma unroll
  for (int kc2 = 0; kc2 < 2; ++kc2) {
    v16h a = load_a_f32(S + kc2 * 32, 96, 32);
#pragma unroll
    for (int nt = 0; nt < 4; ++nt)
      acc2[nt] = wmma16(a, load_b_frag(wh + OFF_WC2 + (kc2 * 4 + nt) * FRAG), acc2[nt]);
  }
  WAIT_LDS();
#pragma unroll
  for (int nt = 0; nt < 4; ++nt) {
    const int col = nt * 16 + m;
    const float bb = bc2[col];
#pragma unroll
    for (int r = 0; r < 8; ++r)
      S[(r + hi8) * 96 + col] = fmaxf(acc2[nt][r] + bb, 0.0f);
  }
  WAIT_LDS();

  // logits = x2 @ Wc3 + bc3 (N padded 8->16)
  v8f cl = {};
#pragma unroll
  for (int kc2 = 0; kc2 < 2; ++kc2)
    cl = wmma16(load_a_f32(S + kc2 * 32, 96, 32), load_b_frag(wh + OFF_WC3 + kc2 * FRAG), cl);
  if (m < 8) {
    const float b3 = bc3[m];
#pragma unroll
    for (int r = 0; r < 8; ++r)
      logits[(size_t)(row0 + r + hi8) * 8 + m] = cl[r] + b3;
  }
}

// ---------------- host launcher ----------------
extern "C" void kernel_launch(void* const* d_in, const int* in_sizes, int n_in,
                              void* d_out, int out_size, void* d_ws, size_t ws_size,
                              hipStream_t stream) {
  (void)in_sizes; (void)n_in; (void)out_size; (void)ws_size;
  const float* feats = (const float*)d_in[0];
  const int*   ntypes= (const int*)  d_in[1];
  const float* hemb  = (const float*)d_in[2];
  const int*   src0  = (const int*)  d_in[3];
  const int*   dst0  = (const int*)  d_in[4];
  const int*   src1  = (const int*)  d_in[5];
  const int*   dst1  = (const int*)  d_in[6];
  const float* Wdev  = (const float*)d_in[9];
  const float* bdev  = (const float*)d_in[10];
  const float* Winst = (const float*)d_in[11];
  const float* binst = (const float*)d_in[12];
  const float* Wnet  = (const float*)d_in[13];
  const float* bnet  = (const float*)d_in[14];
  const float* We1   = (const float*)d_in[15];
  const float* be1   = (const float*)d_in[16];
  const float* We2   = (const float*)d_in[17];
  const float* be2   = (const float*)d_in[18];
  const float* Ws0   = (const float*)d_in[19];
  const float* Wn0   = (const float*)d_in[20];
  const float* bs0   = (const float*)d_in[21];
  const float* Ws1   = (const float*)d_in[22];
  const float* Wn1   = (const float*)d_in[23];
  const float* bs1   = (const float*)d_in[24];
  const float* Wc1   = (const float*)d_in[25];
  const float* bc1   = (const float*)d_in[26];
  const float* Wc2   = (const float*)d_in[27];
  const float* bc2   = (const float*)d_in[28];
  const float* Wc3   = (const float*)d_in[29];
  const float* bc3   = (const float*)d_in[30];

  _Float16* wh = (_Float16*)d_ws;
  float* fb   = (float*)((char*)d_ws + WH_BYTES);
  float* h    = fb;
  float* agg0 = h    + (size_t)N0v * 64;
  float* deg0 = agg0 + (size_t)N1v * 64;
  float* h1   = deg0 + (size_t)N1v;
  float* agg1 = h1   + (size_t)N1v * 64;
  float* deg1 = agg1 + (size_t)N2v * 64;

  float* out    = (float*)d_out;
  float* logits = out;                       // N2 x 8
  float* h2     = out + (size_t)N2v * 8;     // N2 x 64

  k_convert<<<1, 256, 0, stream>>>(We1, We2, Wdev, Winst, Wnet,
                                   Ws0, Wn0, Ws1, Wn1, Wc1, Wc2, Wc3, wh);
  const int zcount = N1v * 64 + N1v + N2v * 64 + N2v;   // agg0, deg0, then start of h1; agg1/deg1 not covered
  k_zero<<<4096, 256, 0, stream>>>(agg0, zcount);

  k_node<<<(N0v / 16 + 7) / 8, 256, 0, stream>>>(feats, ntypes, hemb, be1, be2,
                                                 bdev, binst, bnet, wh, h);
  k_agg<<<(E0v + 255) / 256, 256, 0, stream>>>(src0, dst0, h, agg0, deg0, E0v);

  const int t1 = N1v / 16;   // 9375 tiles
  k_gnn<<<(t1 + 8 * GNN_TPW - 1) / (8 * GNN_TPW), 256, 0, stream>>>(
      h, agg0, deg0, wh, OFF_WS0, OFF_WN0, bs0, h1, N1v, 1);

  k_agg<<<(E1v + 255) / 256, 256, 0, stream>>>(src1, dst1, h1, agg1, deg1, E1v);

  const int t2 = N2v / 16;   // 3125 tiles
  k_gnn<<<(t2 + 8 * GNN_TPW - 1) / (8 * GNN_TPW), 256, 0, stream>>>(
      h1, agg1, deg1, wh, OFF_WS1, OFF_WN1, bs1, h2, N2v, 0);

  k_cls<<<(N2v / 16 + 7) / 8, 256, 0, stream>>>(h2, feats, wh, bc1, bc2, bc3, logits);
}
